// AltAttention_74577812128013
// MI455X (gfx1250) — matrix operand order repaired and checked by reading, not compiled, not run on hardware
//
#include <hip/hip_runtime.h>

// ---------------------------------------------------------------------------
// CDNA5 (gfx1250) bf16 WMMA attention pipeline, v5.
// - all matmuls: v_wmma_f32_16x16x32_bf16, fp32 accumulate
// - GEMMs: block-cooperative 128x64 tile, B staged through double-buffered
//   LDS (fetched once per block per k-chunk), A double-buffered in registers
// - attention: flash-style split-K x4 (4 waves/tile, LDS merge),
//   alibi+mask double-buffered a full chunk ahead
// ---------------------------------------------------------------------------

typedef __bf16 bf16;
typedef __attribute__((ext_vector_type(16))) __bf16 v16bf;
typedef __attribute__((ext_vector_type(8)))  __bf16 v8bf;
typedef __attribute__((ext_vector_type(4)))  __bf16 v4bf;
typedef __attribute__((ext_vector_type(8)))  float  v8f;

__device__ __forceinline__ bf16 f2bf(float f) {
  unsigned u = __builtin_bit_cast(unsigned, f);
  u += 0x7FFFu + ((u >> 16) & 1u);            // round-to-nearest-even
  unsigned short s = (unsigned short)(u >> 16);
  return __builtin_bit_cast(bf16, s);
}

// Lane-adjusted base pointers for WMMA fragments (row-major sources).
__device__ __forceinline__ const bf16* laneA(const bf16* src, int ld, int lane) {
  return src + (size_t)(lane & 15) * ld + ((lane & 16) ? 16 : 0);
}
__device__ __forceinline__ const bf16* laneB(const bf16* src, int ld, int lane) {
  return src + (size_t)(lane & 15) * ld + ((lane & 16) ? 16 : 0);
}

__device__ __forceinline__ v16bf ld_fragA(const bf16* p) {  // +0..7, +8..15
  union { v16bf v; v8bf h[2]; } u;
  u.h[0] = *reinterpret_cast<const v8bf*>(p);
  u.h[1] = *reinterpret_cast<const v8bf*>(p + 8);
  return u.v;
}
__device__ __forceinline__ v16bf ld_fragB(const bf16* p) {  // +0..7, +8..15
  union { v16bf v; v8bf h[2]; } u;
  u.h[0] = *reinterpret_cast<const v8bf*>(p);
  u.h[1] = *reinterpret_cast<const v8bf*>(p + 8);
  return u.v;
}

__device__ __forceinline__ v8f wmma_bf16(v16bf a, v16bf b, v8f c) {
  return __builtin_amdgcn_wmma_f32_16x16x32_bf16(false, a, false, b, (short)0, c,
                                                 false, false);
}

// ---------------------------------------------------------------------------
// Kernel 0: fp32 -> bf16 conversion, 4 elements/thread
// ---------------------------------------------------------------------------
__global__ void k_f32_to_bf16_v4(const float4* __restrict__ in,
                                 v4bf* __restrict__ out, int n4) {
  int i = blockIdx.x * blockDim.x + threadIdx.x;
  if (i < n4) {
    float4 f = in[i];
    v4bf o;
    o[0] = f2bf(f.x); o[1] = f2bf(f.y); o[2] = f2bf(f.z); o[3] = f2bf(f.w);
    out[i] = o;
  }
}

// ---------------------------------------------------------------------------
// Kernel 1/4: C[M,Nc] = A[M,1024](bf16) @ W[Nc,1024](bf16)^T (+ bias), fp32 out.
// Block = 4 waves = 128x64 tile sharing one column group.  Per k-chunk the
// 4 KiB B tile is staged once into double-buffered LDS (written at the end of
// the previous chunk), so waves read B via short-latency ds_load_b128 instead
// of a full L2 round-trip per WMMA pair.  A is double-buffered in registers.
// ---------------------------------------------------------------------------
__global__ __launch_bounds__(128)
void k_gemm_bf16(const bf16* __restrict__ A, const bf16* __restrict__ W,
                 const float* __restrict__ bias, float* __restrict__ C,
                 int M, int Nc, int colGroups) {
  constexpr int K = 1024;
  constexpr int LDB = 40;                      // padded LDS row (80 B)
  __shared__ bf16 Bb[2][64][LDB];

  const int tid  = threadIdx.x;
  const int lane = tid & 31;
  const int w    = tid >> 5;
  const int cg       = blockIdx.x % colGroups;
  const int blockRow = blockIdx.x / colGroups;
  const int j0 = cg * 64;
  const int rt = blockRow * 4 + w;             // this wave's 32-row tile

  const bf16* pA = laneA(A + (size_t)(rt * 32) * K, K, lane);

  // staging: thread copies 16 elements (32 B) of the 64x32 B tile
  const int srow = tid >> 1;                   // 0..63
  const int soff = (tid & 1) << 4;             // 0 or 16 elements
  const bf16* Wst = W + (size_t)(j0 + srow) * K + soff;
  bf16* st0 = &Bb[0][srow][soff];
  bf16* st1 = &Bb[1][srow][soff];

  // LDS fragment base for this lane (tile offsets are immediates)
  const bf16* lb0 = &Bb[0][lane & 15][(lane & 16) ? 16 : 0];
  const bf16* lb1 = &Bb[1][lane & 15][(lane & 16) ? 16 : 0];

  v8f acc[2][4] = {};

  // prologue: stage chunk 0 into buffer 0; A fragments for chunk 0
  {
    v8bf t0 = *reinterpret_cast<const v8bf*>(Wst);
    v8bf t1 = *reinterpret_cast<const v8bf*>(Wst + 8);
    *reinterpret_cast<v8bf*>(st0)     = t0;
    *reinterpret_cast<v8bf*>(st0 + 8) = t1;
  }
  v16bf a0 = ld_fragA(pA);
  v16bf a1 = ld_fragA(pA + 16 * K);
  __syncthreads();

#pragma unroll 2
  for (int c = 0; c < 32; ++c) {               // 32 k-chunks
    const int kn = (c + 1) << 5;               // next chunk k
    // issue next-chunk global loads first; they land during the 8 WMMAs
    v8bf t0, t1;
    v16bf na0, na1;
    if (c < 31) {
      t0  = *reinterpret_cast<const v8bf*>(Wst + kn);
      t1  = *reinterpret_cast<const v8bf*>(Wst + kn + 8);
      na0 = ld_fragA(pA + kn);
      na1 = ld_fragA(pA + kn + 16 * K);
    }

    // B fragments from current LDS buffer (short-latency ds_load_b128)
    const bf16* lb = (c & 1) ? lb1 : lb0;
    v16bf b0 = ld_fragB(lb);
    v16bf b1 = ld_fragB(lb + 16 * LDB);
    v16bf b2 = ld_fragB(lb + 32 * LDB);
    v16bf b3 = ld_fragB(lb + 48 * LDB);

    acc[0][0] = wmma_bf16(a0, b0, acc[0][0]);
    acc[1][0] = wmma_bf16(a1, b0, acc[1][0]);
    acc[0][1] = wmma_bf16(a0, b1, acc[0][1]);
    acc[1][1] = wmma_bf16(a1, b1, acc[1][1]);
    acc[0][2] = wmma_bf16(a0, b2, acc[0][2]);
    acc[1][2] = wmma_bf16(a1, b2, acc[1][2]);
    acc[0][3] = wmma_bf16(a0, b3, acc[0][3]);
    acc[1][3] = wmma_bf16(a1, b3, acc[1][3]);

    if (c < 31) {
      bf16* st = (c & 1) ? st0 : st1;          // write the *other* buffer
      *reinterpret_cast<v8bf*>(st)     = t0;
      *reinterpret_cast<v8bf*>(st + 8) = t1;
      a0 = na0; a1 = na1;
    }
    __syncthreads();
  }

  const int col = lane & 15;
  const int rb8 = (lane & 16) ? 8 : 0;
#pragma unroll
  for (int t = 0; t < 4; ++t) {
    float bv = bias ? bias[j0 + t * 16 + col] : 0.f;
#pragma unroll
    for (int s = 0; s < 2; ++s)
#pragma unroll
      for (int r = 0; r < 8; ++r)
        C[(size_t)(rt * 32 + s * 16 + rb8 + r) * Nc + j0 + t * 16 + col] =
            acc[s][t][r] + bv;
  }
}

// ---------------------------------------------------------------------------
// Kernel 2: split qkv[2048,3072] fp32 into L2-normalized Qn,Kn [B*H,N,64] bf16
// and transposed Vt [B*H,64,N] bf16.  One wave per (b,h,n) row.
// ---------------------------------------------------------------------------
__global__ __launch_bounds__(256)
void k_norm_split(const float* __restrict__ qkv, bf16* __restrict__ Qn,
                  bf16* __restrict__ Kn, bf16* __restrict__ Vt) {
  const int lane = threadIdx.x & 31;
  const int wid  = blockIdx.x * (blockDim.x >> 5) + (threadIdx.x >> 5); // 0..32767
  const int n  = wid & 1023;
  const int bh = wid >> 10;
  const int h  = bh & 15;
  const int b  = bh >> 4;
  const float* row = qkv + ((size_t)(b * 1024 + n)) * 3072;
  const int d0 = lane * 2;
  const size_t obase = ((size_t)bh * 1024 + n) * 64 + d0;

  float2 q2 = *reinterpret_cast<const float2*>(row + h * 64 + d0);
  float2 k2 = *reinterpret_cast<const float2*>(row + 1024 + h * 64 + d0);
  float2 v2 = *reinterpret_cast<const float2*>(row + 2048 + h * 64 + d0);

  float sq = q2.x * q2.x + q2.y * q2.y;
  float sk = k2.x * k2.x + k2.y * k2.y;
#pragma unroll
  for (int m = 1; m < 32; m <<= 1) {
    sq += __shfl_xor(sq, m, 32);
    sk += __shfl_xor(sk, m, 32);
  }
  float scq = 1.0f / fmaxf(sqrtf(sq), 1e-12f);
  float sck = 1.0f / fmaxf(sqrtf(sk), 1e-12f);
  Qn[obase]     = f2bf(q2.x * scq);
  Qn[obase + 1] = f2bf(q2.y * scq);
  Kn[obase]     = f2bf(k2.x * sck);
  Kn[obase + 1] = f2bf(k2.y * sck);
  const size_t vb = (size_t)bh * 64 * 1024;
  Vt[vb + (size_t)d0 * 1024 + n]       = f2bf(v2.x);
  Vt[vb + (size_t)(d0 + 1) * 1024 + n] = f2bf(v2.y);
}

// ---------------------------------------------------------------------------
// Kernel 3: flash attention, split-K x4.
// One block = one (b,h, 16-query) tile; wave w owns keys [w*256,(w+1)*256)
// with its own online softmax; partials merged through LDS after a barrier.
// Alibi + mask are double-buffered one chunk ahead.
// ---------------------------------------------------------------------------
__global__ __launch_bounds__(128)
__attribute__((amdgpu_waves_per_eu(2, 4)))
void k_attn(const bf16* __restrict__ Qn, const bf16* __restrict__ Kn,
            const bf16* __restrict__ Vt, const float* __restrict__ alibi,
            const int* __restrict__ mask, const float* __restrict__ lsc,
            bf16* __restrict__ Obf) {
  __shared__ bf16  Pbuf[4][16 * 32];
  __shared__ float Opart[4][16][64];
  __shared__ float Mpart[4][16];
  __shared__ float Lpart[4][16];

  const int lane = threadIdx.x & 31;
  const int w    = threadIdx.x >> 5;
  const int tile = blockIdx.x;           // 0..2047
  const int qt = tile & 63;
  const int bh = tile >> 6;
  const int h  = bh & 15;
  const int b  = bh >> 4;
  const int q0 = qt << 4;
  const int k0 = w << 8;                 // this wave's 256-key range

  const float ls = __expf(fminf(lsc[h], 4.6051702f)); // clamp at log(100)

  const int col = lane & 15;
  const int rb8 = (lane & 16) ? 8 : 0;

  const bf16*  pK  = laneB(Kn + (size_t)bh * 1024 * 64, 64, lane);
  const bf16*  pV  = laneB(Vt + (size_t)bh * 64 * 1024, 1024, lane);
  const float* pAb = alibi + (size_t)bh * 1024 * 1024 +
                     (size_t)(q0 + rb8) * 1024 + col;
  const int*   pmk = mask + b * 1024 + col;

  const bf16* pQ  = laneA(Qn + ((size_t)bh * 1024 + q0) * 64, 64, lane);
  const v16bf aQ0 = ld_fragA(pQ);         // d = 0..31
  const v16bf aQ1 = ld_fragA(pQ + 32);    // d = 32..63

  bf16* P = Pbuf[w];
  const bf16* pP = P + (lane & 15) * 32 + ((lane & 16) ? 16 : 0);

  const v8f vz = {};
  v8f O[4] = {};
  float mrow[8], lrow[8];
#pragma unroll
  for (int r = 0; r < 8; ++r) { mrow[r] = -1e30f; lrow[r] = 0.f; }

  // prologue: alibi + mask for first chunk
  float ab0[8], ab1[8];
  int mw0 = pmk[k0], mw1 = pmk[k0 + 16];
#pragma unroll
  for (int r = 0; r < 8; ++r) {
    ab0[r] = pAb[r * 1024 + k0];
    ab1[r] = pAb[r * 1024 + k0 + 16];
  }

  for (int kc = k0; kc < k0 + 256; kc += 32) {
    const bf16* kK = pK + (size_t)kc * 64;
    const bf16* kV = pV + kc;

    // K fragments
    v16bf bK00 = ld_fragB(kK);
    v16bf bK01 = ld_fragB(kK + 32);
    v16bf bK10 = ld_fragB(kK + 16 * 64);
    v16bf bK11 = ld_fragB(kK + 16 * 64 + 32);

    // prefetch alibi + mask one full chunk ahead (last iter: wrap, harmless)
    const int kn = (kc + 32 < k0 + 256) ? kc + 32 : k0;
    const int nmw0 = pmk[kn];
    const int nmw1 = pmk[kn + 16];
    float nab0[8], nab1[8];
#pragma unroll
    for (int r = 0; r < 8; ++r) {
      nab0[r] = pAb[r * 1024 + kn];
      nab1[r] = pAb[r * 1024 + kn + 16];
    }

    // QK^T
    v8f S0 = wmma_bf16(aQ0, bK00, vz);
    S0     = wmma_bf16(aQ1, bK01, S0);
    v8f S1 = wmma_bf16(aQ0, bK10, vz);
    S1     = wmma_bf16(aQ1, bK11, S1);

    // V fragments, land during softmax
    v16bf bV0 = ld_fragB(kV);
    v16bf bV1 = ld_fragB(kV + 16 * 1024);
    v16bf bV2 = ld_fragB(kV + 32 * 1024);
    v16bf bV3 = ld_fragB(kV + 48 * 1024);

    // online softmax
#pragma unroll
    for (int r = 0; r < 8; ++r) {
      float s0 = S0[r] * ls + ab0[r];
      float s1 = S1[r] * ls + ab1[r];
      if (mw0) s0 = -1e30f;
      if (mw1) s1 = -1e30f;
      float cm = fmaxf(s0, s1);
      cm = fmaxf(cm, __shfl_xor(cm, 1, 32));
      cm = fmaxf(cm, __shfl_xor(cm, 2, 32));
      cm = fmaxf(cm, __shfl_xor(cm, 4, 32));
      cm = fmaxf(cm, __shfl_xor(cm, 8, 32));
      const float nm    = fmaxf(mrow[r], cm);
      const float scale = __expf(mrow[r] - nm);
      const float p0 = __expf(s0 - nm);
      const float p1 = __expf(s1 - nm);
      float ps = p0 + p1;
      ps += __shfl_xor(ps, 1, 32);
      ps += __shfl_xor(ps, 2, 32);
      ps += __shfl_xor(ps, 4, 32);
      ps += __shfl_xor(ps, 8, 32);
      lrow[r] = lrow[r] * scale + ps;
      mrow[r] = nm;
#pragma unroll
      for (int t = 0; t < 4; ++t) O[t][r] *= scale;
      P[(rb8 + r) * 32 + col]      = f2bf(p0);
      P[(rb8 + r) * 32 + col + 16] = f2bf(p1);
    }

    // rotate alibi/mask double buffer
    mw0 = nmw0; mw1 = nmw1;
#pragma unroll
    for (int r = 0; r < 8; ++r) { ab0[r] = nab0[r]; ab1[r] = nab1[r]; }

    // re-layout P through LDS as an A fragment
    asm volatile("s_wait_dscnt 0x0" ::: "memory");
    const v16bf aP = ld_fragA(pP);

    // PV
    O[0] = wmma_bf16(aP, bV0, O[0]);
    O[1] = wmma_bf16(aP, bV1, O[1]);
    O[2] = wmma_bf16(aP, bV2, O[2]);
    O[3] = wmma_bf16(aP, bV3, O[3]);
  }

  // publish partials
#pragma unroll
  for (int t = 0; t < 4; ++t)
#pragma unroll
    for (int r = 0; r < 8; ++r)
      Opart[w][rb8 + r][t * 16 + col] = O[t][r];
#pragma unroll
  for (int r = 0; r < 8; ++r) {   // 16 lanes of each half write same value/addr
    Mpart[w][rb8 + r] = mrow[r];
    Lpart[w][rb8 + r] = lrow[r];
  }
  __syncthreads();

  // merge: wave w recombines columns [w*16, w*16+16); each lane: 1 row x 8 cols
  const int mr = lane & 15;
  const int cb = w * 16 + (lane >> 4) * 8;
  float mm = -1e30f;
  float ew[4];
#pragma unroll
  for (int u = 0; u < 4; ++u) mm = fmaxf(mm, Mpart[u][mr]);
  float L = 0.f;
#pragma unroll
  for (int u = 0; u < 4; ++u) {
    ew[u] = __expf(Mpart[u][mr] - mm);
    L += Lpart[u][mr] * ew[u];
  }
  const float rcp = L > 0.f ? 1.0f / L : 0.f;
  v8bf ov;
#pragma unroll
  for (int cc = 0; cc < 8; ++cc) {
    float s = 0.f;
#pragma unroll
    for (int u = 0; u < 4; ++u) s += Opart[u][mr][cb + cc] * ew[u];
    ov[cc] = f2bf(s * rcp);
  }
  *reinterpret_cast<v8bf*>(Obf + ((size_t)b * 1024 + q0 + mr) * 1024 + h * 64 + cb) = ov;
}

// ---------------------------------------------------------------------------
// Host launch
// ---------------------------------------------------------------------------
extern "C" void kernel_launch(void* const* d_in, const int* in_sizes, int n_in,
                              void* d_out, int out_size, void* d_ws, size_t ws_size,
                              hipStream_t stream) {
  (void)in_sizes; (void)n_in; (void)out_size; (void)ws_size;
  const float* x      = (const float*)d_in[0]; // [2,1024,1024]
  const int*   pmask  = (const int*)  d_in[1]; // [2,1024]
  const float* alibi  = (const float*)d_in[2]; // [2,16,1024,1024]
  const float* qkv_w  = (const float*)d_in[3]; // [3072,1024]
  const float* proj_w = (const float*)d_in[4]; // [1024,1024]
  const float* proj_b = (const float*)d_in[5]; // [1024]
  const float* lsc    = (const float*)d_in[6]; // [16]

  size_t off = 0;
  auto alloc = [&](size_t bytes) -> void* {
    void* p = (char*)d_ws + off;
    off += (bytes + 255) & ~(size_t)255;
    return p;
  };
  bf16*  xb     = (bf16*) alloc((size_t)2048 * 1024 * 2);
  bf16*  wqkvb  = (bf16*) alloc((size_t)3072 * 1024 * 2);
  bf16*  wprojb = (bf16*) alloc((size_t)1024 * 1024 * 2);
  float* qkvf   = (float*)alloc((size_t)2048 * 3072 * 4);
  bf16*  Qn     = (bf16*) alloc((size_t)32 * 1024 * 64 * 2);
  bf16*  Kn     = (bf16*) alloc((size_t)32 * 1024 * 64 * 2);
  bf16*  Vt     = (bf16*) alloc((size_t)32 * 64 * 1024 * 2);
  bf16*  Obf    = (bf16*) alloc((size_t)2048 * 1024 * 2);

  // 0) bf16 conversions (vectorized x4)
  k_f32_to_bf16_v4<<<(524288 + 255) / 256, 256, 0, stream>>>(
      (const float4*)x, (v4bf*)xb, 524288);
  k_f32_to_bf16_v4<<<(786432 + 255) / 256, 256, 0, stream>>>(
      (const float4*)qkv_w, (v4bf*)wqkvb, 786432);
  k_f32_to_bf16_v4<<<(262144 + 255) / 256, 256, 0, stream>>>(
      (const float4*)proj_w, (v4bf*)wprojb, 262144);

  // 1) QKV projection: [2048,3072] = 16 row-blocks(128) x 48 col-groups(64)
  k_gemm_bf16<<<16 * 48, 128, 0, stream>>>(xb, wqkvb, nullptr, qkvf,
                                           2048, 3072, 48);
  // 2) normalize Q,K + relayout V
  k_norm_split<<<32768 / 8, 256, 0, stream>>>(qkvf, Qn, Kn, Vt);

  // 3) flash attention: one block per 16-query tile, split-K x4 (8192 waves)
  k_attn<<<2048, 128, 0, stream>>>(Qn, Kn, Vt, alibi, pmask, lsc, Obf);

  // 4) output projection + bias: 16 row-blocks x 16 col-groups
  k_gemm_bf16<<<16 * 16, 128, 0, stream>>>(Obf, wprojb, proj_b, (float*)d_out,
                                           2048, 1024, 16);
}
